// imageCaptionModel_27968827032012
// MI455X (gfx1250) — matrix operand order repaired and checked by reading, not compiled, not run on hardware
//
#include <hip/hip_runtime.h>
#include <stdint.h>

// ---- model dims ----
#define BB    128
#define TT    25
#define VOCAB 10000
#define EMB   512
#define VGG   4096
#define HID   512

typedef __bf16 bf16_t;
typedef __attribute__((ext_vector_type(16))) __bf16 v16bf;
typedef __attribute__((ext_vector_type(8)))  __bf16 v8bf;
typedef __attribute__((ext_vector_type(8)))  float  v8f;

// ---------------- WMMA helpers ----------------

__device__ __forceinline__ v8f wmma_bf16(v16bf a, v16bf b, v8f c) {
  // D = A(16x32 bf16) * B(32x16 bf16) + C(16x16 f32)
  return __builtin_amdgcn_wmma_f32_16x16x32_bf16(
      /*neg_a=*/false, a, /*neg_b=*/false, b,
      /*c_mod=*/(short)0, c, /*reuse_a=*/false, /*reuse_b=*/false);
}

// A fragment for 16x32 bf16, row-major activation matrix [M, lda].
// Lane layout (ISA 7.12.2): lanes 0-15 row=lane, K = {0..7, 16..23};
// lanes 16-31 row=lane-16, K = {8..15, 24..31}.
__device__ __forceinline__ v16bf load_a16x32(const bf16_t* __restrict__ A, int lda,
                                             int m0, int k0, int r, int half) {
  const bf16_t* p = A + (size_t)(m0 + r) * lda + k0 + half * 8;
  v8bf lo = *(const v8bf*)(p);
  v8bf hi = *(const v8bf*)(p + 16);
  v16bf a;
#pragma unroll
  for (int i = 0; i < 8; ++i) { a[i] = lo[i]; a[i + 8] = hi[i]; }
  return a;
}

// B fragment from pre-packed weights. Packed layout:
//   P[((nt*kblocks + kb)*32 + lane)*16 + e]  with element e of lane l being
//   W[kb*32 + (e<8 ? (l>>4)*8 + e : 16 + (l>>4)*8 + e-8)][nt*16 + (l&15)]  (ISA B 32x16 layout).
// One contiguous 32B load per lane.
__device__ __forceinline__ v16bf load_b_packed(const bf16_t* __restrict__ P, int kblocks,
                                               int nt, int kb, int lane) {
  return *(const v16bf*)(P + ((((size_t)nt * kblocks + kb) * 32 + lane) << 4));
}

__device__ __forceinline__ float sigmoidf_(float x) {
  return 1.0f / (1.0f + __expf(-x));
}
__device__ __forceinline__ float tanhf_(float x) {
  x = fminf(fmaxf(x, -15.0f), 15.0f);
  float e = __expf(-2.0f * x);
  return (1.0f - e) / (1.0f + e);
}

// ---------------- preprocessing kernels ----------------

// Pack fp32 weight [K,N] (row-major) into bf16 WMMA-B fragment order.
__global__ __launch_bounds__(256) void pack_b_kernel(const float* __restrict__ W,
                                                     bf16_t* __restrict__ P,
                                                     int K, int N) {
  size_t idx = (size_t)blockIdx.x * blockDim.x + threadIdx.x;
  size_t total = (size_t)K * N;
  if (idx >= total) return;
  int e    = (int)(idx & 15);
  int lane = (int)((idx >> 4) & 31);
  size_t rest = idx >> 9;           // /512
  int kblocks = K >> 5;
  int kb = (int)(rest % kblocks);
  int nt = (int)(rest / kblocks);
  int c = lane & 15, half = lane >> 4;
  int k = kb * 32 + ((e < 8) ? (half * 8 + e) : (16 + half * 8 + (e - 8)));
  int n = nt * 16 + c;
  P[idx] = (bf16_t)W[(size_t)k * N + n];
}

__global__ __launch_bounds__(256) void convert_f32_bf16_kernel(const float* __restrict__ src,
                                                               bf16_t* __restrict__ dst,
                                                               int n) {
  int i = blockIdx.x * blockDim.x + threadIdx.x;
  if (i < n) dst[i] = (bf16_t)src[i];
}

// Gather token embeddings into bf16 x_all[t][b][e].
__global__ __launch_bounds__(256) void embed_kernel(const int* __restrict__ tokens,
                                                    const float* __restrict__ emb,
                                                    bf16_t* __restrict__ x_all) {
  int t = blockIdx.x % TT;
  int b = blockIdx.x / TT;
  int tok = tokens[b * TT + t];
  const float* src = emb + (size_t)tok * EMB;
  bf16_t* dst = x_all + ((size_t)t * BB + b) * EMB;
  for (int e = threadIdx.x; e < EMB; e += blockDim.x) dst[e] = (bf16_t)src[e];
}

// ---------------- GEMM kernels ----------------
// Block = one 16-wide N tile; 8 waves = the 8 M tiles (M=128).

// h0 = tanh(vgg @ W_in + b_in); replicate into both layer states.
__global__ __launch_bounds__(256) void h0_kernel(const bf16_t* __restrict__ A,   // [128,4096]
                                                 const bf16_t* __restrict__ P,   // packed [512,4096]
                                                 const float* __restrict__ bias,
                                                 float* __restrict__ s0f, float* __restrict__ s1f,
                                                 bf16_t* __restrict__ s0b, bf16_t* __restrict__ s1b) {
  const int lane = threadIdx.x & 31;
  const int mt = threadIdx.x >> 5;
  const int nt = blockIdx.x;
  const int m0 = mt * 16, n0 = nt * 16;
  const int r = lane & 15, half = lane >> 4;

  v8f acc = {};
#pragma unroll 4
  for (int kb = 0; kb < VGG / 32; ++kb) {
    v16bf a = load_a16x32(A, VGG, m0, kb * 32, r, half);
    v16bf b = load_b_packed(P, VGG / 32, nt, kb, lane);
    acc = wmma_bf16(a, b, acc);
  }
  const float bb = bias[n0 + r];
#pragma unroll
  for (int i = 0; i < 8; ++i) {
    const int row = m0 + half * 8 + i;
    const int idx = row * HID + n0 + r;
    const float v = tanhf_(acc[i] + bb);
    s0f[idx] = v; s1f[idx] = v;
    s0b[idx] = (bf16_t)v; s1b[idx] = (bf16_t)v;
  }
}

// Fused update+reset gates: gu = sigmoid([x|h]@Wu+bu), grh = sigmoid([x|h]@Wr+br) * h.
__global__ __launch_bounds__(256) void gru_gates_kernel(
    const bf16_t* __restrict__ Ax, const bf16_t* __restrict__ Ah,  // [128,512] each
    const bf16_t* __restrict__ PU, const bf16_t* __restrict__ PR,  // packed [512,1024]
    const float* __restrict__ bu, const float* __restrict__ br,
    const float* __restrict__ hf,                                   // fp32 h [128,512]
    float* __restrict__ guf, bf16_t* __restrict__ grhb) {
  const int lane = threadIdx.x & 31;
  const int mt = threadIdx.x >> 5;
  const int nt = blockIdx.x;
  const int m0 = mt * 16, n0 = nt * 16;
  const int r = lane & 15, half = lane >> 4;

  v8f accU = {}, accR = {};
#pragma unroll 4
  for (int kb = 0; kb < 32; ++kb) {            // K = 1024, split at K=512
    const bf16_t* As = (kb < 16) ? Ax : Ah;
    const int kk = (kb < 16) ? (kb * 32) : (kb * 32 - 512);
    v16bf a  = load_a16x32(As, HID, m0, kk, r, half);
    v16bf b0 = load_b_packed(PU, 32, nt, kb, lane);
    v16bf b1 = load_b_packed(PR, 32, nt, kb, lane);
    accU = wmma_bf16(a, b0, accU);             // A fragment reused for both gates
    accR = wmma_bf16(a, b1, accR);
  }
  const float bub = bu[n0 + r];
  const float brb = br[n0 + r];
#pragma unroll
  for (int i = 0; i < 8; ++i) {
    const int row = m0 + half * 8 + i;
    const int idx = row * HID + n0 + r;
    const float gu = sigmoidf_(accU[i] + bub);
    const float gr = sigmoidf_(accR[i] + brb);
    guf[idx]  = gu;
    grhb[idx] = (bf16_t)(gr * hf[idx]);
  }
}

// Candidate + state blend: hh = tanh([x|grh]@Wc+bc); h' = gu*h + (1-gu)*hh.
__global__ __launch_bounds__(256) void gru_cand_kernel(
    const bf16_t* __restrict__ Ax, const bf16_t* __restrict__ Agrh,
    const bf16_t* __restrict__ PC, const float* __restrict__ bc,
    const float* __restrict__ guf,
    float* __restrict__ hf,            // in-out fp32 state
    bf16_t* __restrict__ hb_out) {     // bf16 state copy (next-step h / next-layer x)
  const int lane = threadIdx.x & 31;
  const int mt = threadIdx.x >> 5;
  const int nt = blockIdx.x;
  const int m0 = mt * 16, n0 = nt * 16;
  const int r = lane & 15, half = lane >> 4;

  v8f acc = {};
#pragma unroll 4
  for (int kb = 0; kb < 32; ++kb) {
    const bf16_t* As = (kb < 16) ? Ax : Agrh;
    const int kk = (kb < 16) ? (kb * 32) : (kb * 32 - 512);
    v16bf a = load_a16x32(As, HID, m0, kk, r, half);
    v16bf b = load_b_packed(PC, 32, nt, kb, lane);
    acc = wmma_bf16(a, b, acc);
  }
  const float bcb = bc[n0 + r];
#pragma unroll
  for (int i = 0; i < 8; ++i) {
    const int row = m0 + half * 8 + i;
    const int idx = row * HID + n0 + r;
    const float hh = tanhf_(acc[i] + bcb);
    const float gu = guf[idx];
    const float hn = gu * hf[idx] + (1.0f - gu) * hh;
    hf[idx] = hn;
    hb_out[idx] = (bf16_t)hn;
  }
}

// logits[b, t, :] = s1 @ W_out + b_out   (N = 10000 = 625 tiles of 16)
__global__ __launch_bounds__(256) void logits_kernel(const bf16_t* __restrict__ A, // [128,512]
                                                     const bf16_t* __restrict__ P, // packed [10000,512]
                                                     const float* __restrict__ bias,
                                                     float* __restrict__ out, int t) {
  const int lane = threadIdx.x & 31;
  const int mt = threadIdx.x >> 5;
  const int nt = blockIdx.x;                 // 0..624
  const int m0 = mt * 16, n0 = nt * 16;
  const int r = lane & 15, half = lane >> 4;

  v8f acc = {};
#pragma unroll
  for (int kb = 0; kb < 16; ++kb) {
    v16bf a = load_a16x32(A, HID, m0, kb * 32, r, half);
    v16bf b = load_b_packed(P, 16, nt, kb, lane);
    acc = wmma_bf16(a, b, acc);
  }
  const float bb = bias[n0 + r];
#pragma unroll
  for (int i = 0; i < 8; ++i) {
    const int row = m0 + half * 8 + i;
    const size_t o = (size_t)row * (TT * VOCAB) + (size_t)t * VOCAB + n0 + r;
    out[o] = acc[i] + bb;
  }
}

__global__ __launch_bounds__(256) void copy_state_kernel(const float* __restrict__ s0f,
                                                         const float* __restrict__ s1f,
                                                         float* __restrict__ out) {
  int i = blockIdx.x * blockDim.x + threadIdx.x;
  if (i < BB * HID) {
    out[i] = s0f[i];
    out[BB * HID + i] = s1f[i];
  }
}

// ---------------- host launcher ----------------

extern "C" void kernel_launch(void* const* d_in, const int* in_sizes, int n_in,
                              void* d_out, int out_size, void* d_ws, size_t ws_size,
                              hipStream_t stream) {
  (void)in_sizes; (void)n_in; (void)out_size; (void)ws_size;
  const float* vgg   = (const float*)d_in[0];
  const int*   toks  = (const int*)d_in[1];
  // d_in[2] = is_train (unused by reference)
  const float* emb   = (const float*)d_in[3];
  const float* W_in  = (const float*)d_in[4];
  const float* b_in  = (const float*)d_in[5];
  const float* Wu0   = (const float*)d_in[6];
  const float* bu0   = (const float*)d_in[7];
  const float* Wr0   = (const float*)d_in[8];
  const float* br0   = (const float*)d_in[9];
  const float* Wc0   = (const float*)d_in[10];
  const float* bc0   = (const float*)d_in[11];
  const float* Wu1   = (const float*)d_in[12];
  const float* bu1   = (const float*)d_in[13];
  const float* Wr1   = (const float*)d_in[14];
  const float* br1   = (const float*)d_in[15];
  const float* Wc1   = (const float*)d_in[16];
  const float* bc1   = (const float*)d_in[17];
  const float* W_out = (const float*)d_in[18];
  const float* b_out = (const float*)d_in[19];
  float* out = (float*)d_out;

  // workspace carve-up (256B aligned)
  char* ws = (char*)d_ws;
  size_t off = 0;
  auto carve = [&](size_t bytes) -> char* {
    char* p = ws + off;
    off = (off + bytes + 255) & ~(size_t)255;
    return p;
  };
  bf16_t* p_in  = (bf16_t*)carve((size_t)HID * VGG * 2);
  bf16_t* p_u0  = (bf16_t*)carve((size_t)HID * 1024 * 2);
  bf16_t* p_r0  = (bf16_t*)carve((size_t)HID * 1024 * 2);
  bf16_t* p_c0  = (bf16_t*)carve((size_t)HID * 1024 * 2);
  bf16_t* p_u1  = (bf16_t*)carve((size_t)HID * 1024 * 2);
  bf16_t* p_r1  = (bf16_t*)carve((size_t)HID * 1024 * 2);
  bf16_t* p_c1  = (bf16_t*)carve((size_t)HID * 1024 * 2);
  bf16_t* p_out = (bf16_t*)carve((size_t)VOCAB * HID * 2);
  bf16_t* vggb  = (bf16_t*)carve((size_t)BB * VGG * 2);
  bf16_t* xall  = (bf16_t*)carve((size_t)TT * BB * EMB * 2);
  float*  s0f   = (float*)carve((size_t)BB * HID * 4);
  float*  s1f   = (float*)carve((size_t)BB * HID * 4);
  float*  guf   = (float*)carve((size_t)BB * HID * 4);
  bf16_t* grhb  = (bf16_t*)carve((size_t)BB * HID * 2);
  bf16_t* s0b   = (bf16_t*)carve((size_t)BB * HID * 2);
  bf16_t* s1b   = (bf16_t*)carve((size_t)BB * HID * 2);

  auto pack = [&](const float* W, bf16_t* P, int K, int N) {
    size_t total = (size_t)K * N;
    pack_b_kernel<<<dim3((unsigned)((total + 255) / 256)), dim3(256), 0, stream>>>(W, P, K, N);
  };
  pack(W_in, p_in, VGG, HID);
  pack(Wu0, p_u0, 1024, HID);
  pack(Wr0, p_r0, 1024, HID);
  pack(Wc0, p_c0, 1024, HID);
  pack(Wu1, p_u1, 1024, HID);
  pack(Wr1, p_r1, 1024, HID);
  pack(Wc1, p_c1, 1024, HID);
  pack(W_out, p_out, HID, VOCAB);

  convert_f32_bf16_kernel<<<dim3((BB * VGG + 255) / 256), dim3(256), 0, stream>>>(
      vgg, vggb, BB * VGG);
  embed_kernel<<<dim3(BB * TT), dim3(256), 0, stream>>>(toks, emb, xall);

  // h0 = tanh(vgg @ W_in + b_in) -> both layer states
  h0_kernel<<<dim3(HID / 16), dim3(256), 0, stream>>>(vggb, p_in, b_in, s0f, s1f, s0b, s1b);

  for (int t = 0; t < TT; ++t) {
    const bf16_t* xt = xall + (size_t)t * BB * EMB;
    // layer 0
    gru_gates_kernel<<<dim3(HID / 16), dim3(256), 0, stream>>>(
        xt, s0b, p_u0, p_r0, bu0, br0, s0f, guf, grhb);
    gru_cand_kernel<<<dim3(HID / 16), dim3(256), 0, stream>>>(
        xt, grhb, p_c0, bc0, guf, s0f, s0b);
    // layer 1 (x = new s0)
    gru_gates_kernel<<<dim3(HID / 16), dim3(256), 0, stream>>>(
        s0b, s1b, p_u1, p_r1, bu1, br1, s1f, guf, grhb);
    gru_cand_kernel<<<dim3(HID / 16), dim3(256), 0, stream>>>(
        s0b, grhb, p_c1, bc1, guf, s1f, s1b);
    // logits
    logits_kernel<<<dim3(VOCAB / 16), dim3(256), 0, stream>>>(s1b, p_out, b_out, out, t);
  }

  copy_state_kernel<<<dim3((BB * HID + 255) / 256), dim3(256), 0, stream>>>(
      s0f, s1f, out + (size_t)BB * TT * VOCAB);
}
